// GraphEncoder_33002528703268
// MI455X (gfx1250) — matrix operand order repaired and checked by reading, not compiled, not run on hardware
//
#include <hip/hip_runtime.h>

#define NN    16384          // nodes
#define EE    65536          // edges (before self loops)
#define ETOT  (EE + NN)      // 81920
#define GG    64             // graphs
#define IND   384            // input dim
#define HIDD  512            // hidden per head
#define NH    4              // heads
#define HH    2048           // heads*hidden
#define TOKD  8              // output tokens
#define SLOPE 0.2f

// gfx1250 async global->LDS staging (ISA §15.18, GLOBAL_LOAD_ASYNC_TO_LDS_B128,
// tracked by ASYNCcnt). Set to 0 to fall back to ds_store staging.
#define USE_ASYNC_LDS 1

typedef __attribute__((ext_vector_type(16))) __bf16 v16bf;
typedef __attribute__((ext_vector_type(8)))  float  v8f;

__device__ __forceinline__ unsigned short f2bf(float f) {
  unsigned int b = __float_as_uint(f);
  unsigned int r = (b + 0x7FFFu + ((b >> 16) & 1u)) >> 16;  // RNE
  return (unsigned short)r;
}

__device__ __forceinline__ void atomicMaxF(float* a, float v) {
  if (v >= 0.0f) atomicMax((int*)a, __float_as_int(v));
  else           atomicMin((unsigned int*)a, __float_as_uint(v));
}

// 16B global -> LDS copy. Async path: VDST VGPR holds the LDS byte offset
// (= low 32 bits of the flat shared pointer), address VGPR pair holds the
// 64-bit global address.
__device__ __forceinline__ void lds_copy16(const unsigned short* __restrict__ g,
                                           unsigned int* __restrict__ l) {
#if USE_ASYNC_LDS
  const unsigned int lds_off = (unsigned int)(unsigned long long)l;
  asm volatile("global_load_async_to_lds_b128 %0, %1, off"
               :: "v"(lds_off), "v"(g)
               : "memory");
#else
  *(uint4*)l = *(const uint4*)g;
#endif
}

__device__ __forceinline__ void wait_async_prev() {   // previous tile landed
#if USE_ASYNC_LDS
  asm volatile("s_wait_asynccnt 0x4" ::: "memory");
#endif
}
__device__ __forceinline__ void wait_async_all() {    // drain
#if USE_ASYNC_LDS
  asm volatile("s_wait_asynccnt 0x0" ::: "memory");
#endif
}

// ---------------------------------------------------------------- conversions
__global__ __launch_bounds__(256)
void k_f32_to_bf16(const float* __restrict__ in, unsigned short* __restrict__ out, int n) {
  int i = blockIdx.x * 256 + threadIdx.x;
  if (i < n) out[i] = f2bf(in[i]);
}

// in: [K,N] f32 row-major  ->  out: [N,K] bf16 (transposed, K contiguous)
__global__ __launch_bounds__(256)
void k_f32_to_bf16_t(const float* __restrict__ in, unsigned short* __restrict__ out,
                     int K, int N) {
  int i = blockIdx.x * 256 + threadIdx.x;
  if (i < K * N) {
    int k = i / N, n = i - k * N;
    out[(size_t)n * K + k] = f2bf(in[i]);
  }
}

__global__ __launch_bounds__(256)
void k_fill(float* __restrict__ p, float v, int n) {
  int i = blockIdx.x * 256 + threadIdx.x;
  if (i < n) p[i] = v;
}

// ---------------------------------------------------------------- WMMA GEMM
// C[M,N] = A[M,K](bf16) * Bt[N,K](bf16)^T + bias, optional ReLU.
// 256 threads = 8 waves (4x2 wave grid); block tile 128x128; wave tile 32x64;
// K-step 32; double-buffered LDS fed by async global->LDS loads.
// Requires: N % 128 == 0, K % 32 == 0 (true for all call sites).
__device__ __forceinline__ void gemm_stage(const unsigned short* __restrict__ A,
                                           const unsigned short* __restrict__ Bt,
                                           unsigned int* __restrict__ dA,
                                           unsigned int* __restrict__ dB,
                                           int m0, int n0, int M, int K, int k0,
                                           int tid) {
#pragma unroll
  for (int rr = 0; rr < 2; ++rr) {
    const int row = (tid >> 2) + rr * 64;        // 0..127
    const int cu  = (tid & 3) << 2;              // dword col 0,4,8,12
    int gr = m0 + row; gr = (gr < M) ? gr : (M - 1);   // clamp, no divergence
    lds_copy16(A + (size_t)gr * K + k0 + (cu << 1), dA + row * 20 + cu);
    const int gn = n0 + row;                     // N multiple of 128
    lds_copy16(Bt + (size_t)gn * K + k0 + (cu << 1), dB + row * 20 + cu);
  }
}

__global__ __launch_bounds__(256)
void k_gemm_wmma(const unsigned short* __restrict__ A,
                 const unsigned short* __restrict__ Bt,
                 const float* __restrict__ bias,
                 float* __restrict__ C,
                 int M, int N, int K, int relu) {
  __shared__ unsigned int sA[2][128 * 20];  // 128 rows x 16 dwords (+4 pad)
  __shared__ unsigned int sB[2][128 * 20];  // 128 cols x 16 dwords (+4 pad)

  const int tid  = threadIdx.x;
  const int wave = tid >> 5;
  const int lane = tid & 31;
  const int half = lane >> 4;
  const int l16  = lane & 15;
  const int wm   = wave >> 1;   // 0..3 -> 32-row strip
  const int wn   = wave & 1;    // 0..1 -> 64-col strip
  const int m0 = blockIdx.y * 128;
  const int n0 = blockIdx.x * 128;

  v8f acc[2][4] = {};

  const int KT = K >> 5;
  gemm_stage(A, Bt, sA[0], sB[0], m0, n0, M, K, 0, tid);   // prologue: tile 0

  for (int kt = 0; kt < KT; ++kt) {
    const int cur = kt & 1;
    if (kt + 1 < KT) {
      gemm_stage(A, Bt, sA[cur ^ 1], sB[cur ^ 1], m0, n0, M, K, (kt + 1) << 5, tid);
      wait_async_prev();   // 4 newer in flight; ensure tile kt's 4 landed
    } else {
      wait_async_all();
    }
    __syncthreads();       // tile kt visible to all waves

    // A fragments: 16x32, row = wm*32 + ms*16 + l16; elements 0-7 -> K=half*8+i,
    // elements 8-15 -> K=16+half*8+i (ISA 16-bit A layout)
    union { v16bf v; unsigned int u[8]; } af[2];
#pragma unroll
    for (int ms = 0; ms < 2; ++ms) {
      const unsigned int* pa = &sA[cur][(wm * 32 + ms * 16 + l16) * 20];
#pragma unroll
      for (int j = 0; j < 4; ++j) {
        af[ms].u[j]     = pa[half * 4 + j];
        af[ms].u[4 + j] = pa[8 + half * 4 + j];
      }
    }
    // B fragments (all 4 materialized up front so their ds_loads overlap and
    // the 8 WMMAs below issue back-to-back): 32x16, col = wn*64 + ns*16 + l16,
    // elements 0-7 -> K=half*8+i, elements 8-15 -> K=16+half*8+i
    union { v16bf v; unsigned int u[8]; } bfm[4];
#pragma unroll
    for (int ns = 0; ns < 4; ++ns) {
      const unsigned int* pb = &sB[cur][(wn * 64 + ns * 16 + l16) * 20];
#pragma unroll
      for (int j = 0; j < 4; ++j) { bfm[ns].u[j] = pb[half * 4 + j]; bfm[ns].u[4 + j] = pb[8 + half * 4 + j]; }
    }
#pragma unroll
    for (int ns = 0; ns < 4; ++ns) {
      acc[0][ns] = __builtin_amdgcn_wmma_f32_16x16x32_bf16(
          false, af[0].v, false, bfm[ns].v, (short)0, acc[0][ns], false, false);
      acc[1][ns] = __builtin_amdgcn_wmma_f32_16x16x32_bf16(
          false, af[1].v, false, bfm[ns].v, (short)0, acc[1][ns], false, false);
    }
    __syncthreads();       // done reading buf[cur]; next iter may overwrite it
  }

  // store: C/D layout -> VGPR r holds row (half*8 + r), col = l16
#pragma unroll
  for (int ms = 0; ms < 2; ++ms) {
    const int rowb = m0 + wm * 32 + ms * 16 + half * 8;
#pragma unroll
    for (int ns = 0; ns < 4; ++ns) {
      int col  = n0 + wn * 64 + ns * 16 + l16;
      float bv = bias ? bias[col] : 0.0f;
#pragma unroll
      for (int r = 0; r < 8; ++r) {
        int row = rowb + r;
        if (row < M) {
          float v = acc[ms][ns][r] + bv;
          if (relu) v = fmaxf(v, 0.0f);
          C[(size_t)row * N + col] = v;
        }
      }
    }
  }
}

// ---------------------------------------------------------------- edge stage
// one wave32 per (edge, head); lane covers 16 channels
__global__ __launch_bounds__(256)
void k_edge_scores(const float* __restrict__ xl, const float* __restrict__ xr,
                   const int* __restrict__ ei, const float* __restrict__ att,
                   float* __restrict__ esc, float* __restrict__ emax) {
  int task = blockIdx.x * 8 + (threadIdx.x >> 5);
  int lane = threadIdx.x & 31;
  int e = task >> 2, h = task & 3;
  int src, dst;
  if (e < EE) { src = ei[e]; dst = ei[EE + e]; }
  else        { src = e - EE; dst = src; }
  const float4* pl = (const float4*)(xl + (size_t)src * HH + h * HIDD + lane * 16);
  const float4* pr = (const float4*)(xr + (size_t)dst * HH + h * HIDD + lane * 16);
  const float4* pa = (const float4*)(att + h * HIDD + lane * 16);
  float s = 0.0f;
#pragma unroll
  for (int j = 0; j < 4; ++j) {
    float4 a = pl[j], b = pr[j], w = pa[j];
    float m;
    m = a.x + b.x; s += (m > 0.0f ? m : m * SLOPE) * w.x;
    m = a.y + b.y; s += (m > 0.0f ? m : m * SLOPE) * w.y;
    m = a.z + b.z; s += (m > 0.0f ? m : m * SLOPE) * w.z;
    m = a.w + b.w; s += (m > 0.0f ? m : m * SLOPE) * w.w;
  }
#pragma unroll
  for (int off = 16; off > 0; off >>= 1) s += __shfl_down(s, off, 32);
  if (lane == 0) {
    esc[task] = s;
    atomicMaxF(&emax[dst * NH + h], s);
  }
}

__global__ __launch_bounds__(256)
void k_edge_exp(float* __restrict__ esc, const float* __restrict__ emax,
                float* __restrict__ den, const int* __restrict__ ei) {
  int t = blockIdx.x * 256 + threadIdx.x;
  if (t >= ETOT * NH) return;
  int e = t >> 2, h = t & 3;
  int dst = (e < EE) ? ei[EE + e] : (e - EE);
  float ex = __expf(esc[t] - emax[dst * NH + h]);
  esc[t] = ex;
  atomicAdd(&den[dst * NH + h], ex);
}

__global__ __launch_bounds__(256)
void k_edge_agg(const float* __restrict__ xl, const float* __restrict__ esc,
                const float* __restrict__ den, const int* __restrict__ ei,
                float* __restrict__ out) {
  int task = blockIdx.x * 8 + (threadIdx.x >> 5);
  int lane = threadIdx.x & 31;
  int e = task >> 2, h = task & 3;
  int src, dst;
  if (e < EE) { src = ei[e]; dst = ei[EE + e]; }
  else        { src = e - EE; dst = src; }
  float alpha = esc[task] / den[dst * NH + h];
  const float4* pl = (const float4*)(xl + (size_t)src * HH + h * HIDD + lane * 16);
  float* po = out + (size_t)dst * HH + h * HIDD + lane * 16;
#pragma unroll
  for (int j = 0; j < 4; ++j) {
    float4 a = pl[j];
    atomicAdd(po + 4 * j + 0, alpha * a.x);
    atomicAdd(po + 4 * j + 1, alpha * a.y);
    atomicAdd(po + 4 * j + 2, alpha * a.z);
    atomicAdd(po + 4 * j + 3, alpha * a.w);
  }
}

__global__ __launch_bounds__(256)
void k_bias_relu(float* __restrict__ x, const float* __restrict__ b, int n) {
  int i = blockIdx.x * 256 + threadIdx.x;
  if (i < n) x[i] = fmaxf(x[i] + b[i & (HH - 1)], 0.0f);
}

// ---------------------------------------------------------------- pooling
__global__ __launch_bounds__(256)
void k_count(const int* __restrict__ batch, float* __restrict__ cnt) {
  int i = blockIdx.x * 256 + threadIdx.x;
  if (i < NN) atomicAdd(&cnt[batch[i]], 1.0f);
}

__global__ __launch_bounds__(256)
void k_pool(const float* __restrict__ x, const int* __restrict__ batch,
            float* __restrict__ pooled) {
  int i = blockIdx.x * 256 + threadIdx.x;
  if (i < NN * HH) {
    int nidx = i >> 11;          // / 2048
    int c    = i & (HH - 1);
    atomicAdd(&pooled[batch[nidx] * HH + c], x[i]);
  }
}

__global__ __launch_bounds__(256)
void k_pool_div(float* __restrict__ pooled, const float* __restrict__ cnt) {
  int i = blockIdx.x * 256 + threadIdx.x;
  if (i < GG * HH) pooled[i] = pooled[i] / fmaxf(cnt[i >> 11], 1.0f);
}

// ---------------------------------------------------------------- launcher
extern "C" void kernel_launch(void* const* d_in, const int* in_sizes, int n_in,
                              void* d_out, int out_size, void* d_ws, size_t ws_size,
                              hipStream_t stream) {
  (void)in_sizes; (void)n_in; (void)out_size; (void)ws_size;
  const float* x     = (const float*)d_in[0];
  const int*   ei    = (const int*)d_in[1];
  const int*   batch = (const int*)d_in[2];
  const float *Wl[3], *bl[3], *Wr[3], *br[3], *att[3], *lbias[3];
  for (int l = 0; l < 3; ++l) {
    int b = 3 + l * 6;
    Wl[l]    = (const float*)d_in[b + 0];
    bl[l]    = (const float*)d_in[b + 1];
    Wr[l]    = (const float*)d_in[b + 2];
    br[l]    = (const float*)d_in[b + 3];
    att[l]   = (const float*)d_in[b + 4];
    lbias[l] = (const float*)d_in[b + 5];
  }
  const float* W1 = (const float*)d_in[21];
  const float* b1 = (const float*)d_in[22];
  const float* W2 = (const float*)d_in[23];
  const float* b2 = (const float*)d_in[24];
  float* out = (float*)d_out;

  char* wsp = (char*)d_ws;
  auto carve = [&](size_t bytes) -> char* {
    char* p = wsp; wsp += (bytes + 255) & ~(size_t)255; return p;
  };
  float*          bufA   = (float*)carve((size_t)NN * HH * 4);
  float*          bufB   = (float*)carve((size_t)NN * HH * 4);
  float*          bufC   = (float*)carve((size_t)NN * HH * 4);
  unsigned short* abf    = (unsigned short*)carve((size_t)NN * HH * 2);
  unsigned short* wbf    = (unsigned short*)carve((size_t)HH * HH * 2);
  float*          esc    = (float*)carve((size_t)ETOT * NH * 4);
  float*          emax   = (float*)carve((size_t)NN * NH * 4);
  float*          eden   = (float*)carve((size_t)NN * NH * 4);
  float*          pooled = (float*)carve((size_t)GG * HH * 4);
  float*          hbuf   = (float*)carve((size_t)GG * HIDD * 4);
  float*          cnt    = (float*)carve((size_t)GG * 4);

  float* bufs[3] = {bufA, bufB, bufC};
  const int xli[3] = {0, 2, 1};   // xl buffer per layer
  const int xri[3] = {1, 0, 2};   // xr buffer (reused as aggregation output)

  const dim3 T(256);
  const float* cur = x;
  int D = IND;
  for (int l = 0; l < 3; ++l) {
    float* xl = bufs[xli[l]];
    float* xo = bufs[xri[l]];
    int nA = NN * D, nW = D * HH;
    k_f32_to_bf16<<<dim3((nA + 255) / 256), T, 0, stream>>>(cur, abf, nA);
    // xl = cur @ Wl + bl
    k_f32_to_bf16_t<<<dim3((nW + 255) / 256), T, 0, stream>>>(Wl[l], wbf, D, HH);
    k_gemm_wmma<<<dim3(HH / 128, NN / 128), T, 0, stream>>>(abf, wbf, bl[l], xl, NN, HH, D, 0);
    // xr = cur @ Wr + br
    k_f32_to_bf16_t<<<dim3((nW + 255) / 256), T, 0, stream>>>(Wr[l], wbf, D, HH);
    k_gemm_wmma<<<dim3(HH / 128, NN / 128), T, 0, stream>>>(abf, wbf, br[l], xo, NN, HH, D, 0);
    // segment softmax attention + aggregation
    k_fill<<<dim3((NN * NH + 255) / 256), T, 0, stream>>>(emax, -3.0e38f, NN * NH);
    k_fill<<<dim3((NN * NH + 255) / 256), T, 0, stream>>>(eden, 0.0f, NN * NH);
    k_edge_scores<<<dim3(ETOT * NH / 8), T, 0, stream>>>(xl, xo, ei, att[l], esc, emax);
    k_fill<<<dim3((NN * HH + 255) / 256), T, 0, stream>>>(xo, 0.0f, NN * HH);  // xr dead -> out
    k_edge_exp<<<dim3((ETOT * NH + 255) / 256), T, 0, stream>>>(esc, emax, eden, ei);
    k_edge_agg<<<dim3(ETOT * NH / 8), T, 0, stream>>>(xl, esc, eden, ei, xo);
    k_bias_relu<<<dim3((NN * HH + 255) / 256), T, 0, stream>>>(xo, lbias[l], NN * HH);
    cur = xo; D = HH;
  }

  // mean pool per graph
  k_fill<<<dim3(1), dim3(64), 0, stream>>>(cnt, 0.0f, GG);
  k_fill<<<dim3((GG * HH + 255) / 256), T, 0, stream>>>(pooled, 0.0f, GG * HH);
  k_count<<<dim3(NN / 256), T, 0, stream>>>(batch, cnt);
  k_pool<<<dim3((NN * HH + 255) / 256), T, 0, stream>>>(cur, batch, pooled);
  k_pool_div<<<dim3((GG * HH + 255) / 256), T, 0, stream>>>(pooled, cnt);

  // MLP head: h = relu(pooled @ W1 + b1); out = h @ W2 + b2
  k_f32_to_bf16<<<dim3((GG * HH + 255) / 256), T, 0, stream>>>(pooled, abf, GG * HH);
  k_f32_to_bf16_t<<<dim3((HH * HIDD + 255) / 256), T, 0, stream>>>(W1, wbf, HH, HIDD);
  k_gemm_wmma<<<dim3(HIDD / 128, 1), T, 0, stream>>>(abf, wbf, b1, hbuf, GG, HIDD, HH, 1);
  k_f32_to_bf16<<<dim3((GG * HIDD + 255) / 256), T, 0, stream>>>(hbuf, abf, GG * HIDD);
  k_f32_to_bf16_t<<<dim3((HIDD * (TOKD * IND) + 255) / 256), T, 0, stream>>>(W2, wbf, HIDD, TOKD * IND);
  k_gemm_wmma<<<dim3((TOKD * IND) / 128, 1), T, 0, stream>>>(abf, wbf, b2, out, GG, TOKD * IND, HIDD, 0);
}
